// RGCNQuant_65687229826000
// MI455X (gfx1250) — 8-bit matrix operand order repaired and checked by reading, compiled, not run on hardware
//
#include <hip/hip_runtime.h>
#include <stdint.h>

#define NPAPER 200000
#define NAUTH  100000
#define NEDGE  1500000
#define CINF   128
#define HDIM   128
#define COUTD  64

typedef __attribute__((ext_vector_type(8))) int v8i;

__device__ __forceinline__ unsigned f2key(float f) {
    unsigned u = __float_as_uint(f);
    return (u & 0x80000000u) ? ~u : (u | 0x80000000u);
}
__device__ __forceinline__ float key2f(unsigned k) {
    return (k & 0x80000000u) ? __uint_as_float(k & 0x7FFFFFFFu)
                             : __uint_as_float(~k);
}
__device__ __forceinline__ void wave_minmax(unsigned &kmn, unsigned &kmx) {
#pragma unroll
    for (int off = 16; off > 0; off >>= 1) {
        unsigned a = (unsigned)__shfl_xor((int)kmn, off, 32);
        unsigned b = (unsigned)__shfl_xor((int)kmx, off, 32);
        kmn = (a < kmn) ? a : kmn;
        kmx = (b > kmx) ? b : kmx;
    }
}

__global__ void init_slots_kernel(unsigned *mnmx, int nslots) {
    int i = threadIdx.x;
    if (i < nslots) { mnmx[2 * i] = 0xFFFFFFFFu; mnmx[2 * i + 1] = 0u; }
}

__global__ void zero_kernel(float *p, int n) {
    int i = blockIdx.x * blockDim.x + threadIdx.x;
    if (i < n) p[i] = 0.0f;
}

__global__ __launch_bounds__(256)
void minmax_kernel(const float *__restrict__ x, int n, unsigned *__restrict__ mnmx) {
    unsigned kmn = 0xFFFFFFFFu, kmx = 0u;
    for (int i = blockIdx.x * blockDim.x + threadIdx.x; i < n;
         i += gridDim.x * blockDim.x) {
        unsigned k = f2key(x[i]);
        kmn = (k < kmn) ? k : kmn;
        kmx = (k > kmx) ? k : kmx;
    }
    wave_minmax(kmn, kmx);
    if ((threadIdx.x & 31) == 0) {
        atomicMin(&mnmx[0], kmn);
        atomicMax(&mnmx[1], kmx);
    }
}

__global__ __launch_bounds__(256)
void finalize_mean_kernel(float *__restrict__ s, const float *__restrict__ deg,
                          int nrows, unsigned *__restrict__ mnmx) {
    int n = nrows * CINF;
    unsigned kmn = 0xFFFFFFFFu, kmx = 0u;
    for (int i = blockIdx.x * blockDim.x + threadIdx.x; i < n;
         i += gridDim.x * blockDim.x) {
        int row = i >> 7;
        float d = deg[row];
        d = (d > 1.0f) ? d : 1.0f;
        float v = s[i] / d;
        s[i] = v;
        unsigned k = f2key(v);
        kmn = (k < kmn) ? k : kmn;
        kmx = (k > kmx) ? k : kmx;
    }
    wave_minmax(kmn, kmx);
    if ((threadIdx.x & 31) == 0) {
        atomicMin(&mnmx[0], kmn);
        atomicMax(&mnmx[1], kmx);
    }
}

__global__ __launch_bounds__(256)
void quantize_act_kernel(const float *__restrict__ x, int n,
                         const unsigned *__restrict__ mnmx,
                         unsigned char *__restrict__ q, float *__restrict__ par) {
    float mn = key2f(mnmx[0]);
    float mx = key2f(mnmx[1]);
    float scale = (mx > mn) ? (mx - mn) * (1.0f / 15.0f) : 1.0f;
    if (blockIdx.x == 0 && threadIdx.x == 0) { par[0] = mn; par[1] = scale; }
    for (int i = blockIdx.x * blockDim.x + threadIdx.x; i < n;
         i += gridDim.x * blockDim.x) {
        float t = rintf((x[i] - mn) / scale);
        t = (t < 0.0f) ? 0.0f : ((t > 15.0f) ? 15.0f : t);
        q[i] = (unsigned char)t;
    }
}

__global__ __launch_bounds__(256)
void quantize_wt_kernel(const float *__restrict__ w, int rows,
                        signed char *__restrict__ qw, float *__restrict__ bscale,
                        int *__restrict__ rowsum) {
    __shared__ float red[256];
    int n = rows * CINF;
    float am = 0.0f;
    for (int i = threadIdx.x; i < n; i += 256) am = fmaxf(am, fabsf(w[i]));
    red[threadIdx.x] = am;
    __syncthreads();
    for (int s = 128; s > 0; s >>= 1) {
        if (threadIdx.x < s) red[threadIdx.x] = fmaxf(red[threadIdx.x], red[threadIdx.x + s]);
        __syncthreads();
    }
    float amax = red[0];
    float sc = (amax > 0.0f) ? amax * (1.0f / 7.0f) : 1.0f;
    if (threadIdx.x == 0) bscale[0] = sc;
    for (int i = threadIdx.x; i < rows; i += 256) rowsum[i] = 0;
    __syncthreads();
    for (int i = threadIdx.x; i < n; i += 256) {
        int qv = (int)rintf(w[i] / sc);
        qv = (qv < -8) ? -8 : ((qv > 7) ? 7 : qv);
        qw[i] = (signed char)qv;
        atomicAdd(&rowsum[i >> 7], qv);
    }
}

__global__ __launch_bounds__(256)
void edge_scatter_kernel(const float *__restrict__ x, const int *__restrict__ src,
                         const int *__restrict__ dst, float *__restrict__ s,
                         float *__restrict__ deg, int ne) {
    int tid = blockIdx.x * blockDim.x + threadIdx.x;
    int e = tid >> 5;
    if (e >= ne) return;
    int lane = tid & 31;
    int si = src[e], di = dst[e];
    const float4 v = reinterpret_cast<const float4 *>(x + (size_t)si * CINF)[lane];
    float *o = s + (size_t)di * CINF + lane * 4;
    atomicAdd(o + 0, v.x);
    atomicAdd(o + 1, v.y);
    atomicAdd(o + 2, v.z);
    atomicAdd(o + 3, v.w);
    if (lane == 0) atomicAdd(&deg[di], 1.0f);
}

template <int ACC, int RELU>
__global__ __launch_bounds__(128)
void gemm_q8_kernel(const unsigned char *__restrict__ A, const float *__restrict__ apar,
                    const signed char *__restrict__ B, const float *__restrict__ bscale,
                    const int *__restrict__ browsum, float *__restrict__ out,
                    int M, int N) {
    const int lane = threadIdx.x & 31;
    const int wave = threadIdx.x >> 5;
    const int h = lane >> 4;
    const int r = lane & 15;
    const int row0 = (blockIdx.x * 4 + wave) * 16;
    const int col0 = blockIdx.y * 64;
    if (row0 >= M) return;

    const unsigned *A32 = reinterpret_cast<const unsigned *>(A) + (size_t)(row0 + r) * (CINF / 4);

    v8i a0, a1;
#pragma unroll
    for (int v = 0; v < 8; ++v) {
        int ka = (v >> 1) * 4 + (v & 1) + h * 2;
        a0[v] = (int)A32[ka];
        a1[v] = (int)A32[ka + 16];
    }

    v8i b0[4], b1[4];
#pragma unroll
    for (int t = 0; t < 4; ++t) {
        const unsigned *B32 =
            reinterpret_cast<const unsigned *>(B) + (size_t)(col0 + t * 16 + r) * (CINF / 4);
#pragma unroll
        for (int v = 0; v < 8; ++v) {
            int kb = (v >> 1) * 4 + (v & 1) + h * 2;
            b0[t][v] = (int)B32[kb];
            b1[t][v] = (int)B32[kb + 16];
        }
    }

    v8i c[4] = {v8i{}, v8i{}, v8i{}, v8i{}};
#pragma unroll
    for (int t = 0; t < 4; ++t)
        c[t] = __builtin_amdgcn_wmma_i32_16x16x64_iu8(false, a0, true, b0[t], c[t], false, false);
#pragma unroll
    for (int t = 0; t < 4; ++t)
        c[t] = __builtin_amdgcn_wmma_i32_16x16x64_iu8(false, a1, true, b1[t], c[t], false, false);

    const float mn = apar[0], sa = apar[1];
    const float sb = bscale[0];
    const float sab = sa * sb;
#pragma unroll
    for (int t = 0; t < 4; ++t) {
        const int col = col0 + t * 16 + r;
        const float corr = mn * sb * (float)browsum[col];
#pragma unroll
        for (int v = 0; v < 8; ++v) {
            int row = row0 + v + h * 8;
            float val = sab * (float)c[t][v] + corr;
            size_t idx = (size_t)row * N + col;
            if (ACC) val += out[idx];
            if (RELU) val = fmaxf(val, 0.0f);
            out[idx] = val;
        }
    }
}

static inline void *wsoff(void *ws, size_t &off, size_t bytes) {
    void *p = (char *)ws + off;
    off += (bytes + 255) & ~(size_t)255;
    return p;
}

extern "C" void kernel_launch(void *const *d_in, const int *in_sizes, int n_in,
                              void *d_out, int out_size, void *d_ws, size_t ws_size,
                              hipStream_t stream) {
    const float *x_p = (const float *)d_in[0];
    const float *x_a = (const float *)d_in[1];
    const float *W0[5] = {(const float *)d_in[2], (const float *)d_in[3],
                          (const float *)d_in[4], (const float *)d_in[5],
                          (const float *)d_in[6]};
    const float *W1[5] = {(const float *)d_in[7], (const float *)d_in[8],
                          (const float *)d_in[9], (const float *)d_in[10],
                          (const float *)d_in[11]};
    const int *cites_src = (const int *)d_in[12];
    const int *cites_dst = (const int *)d_in[13];
    const int *writes_src = (const int *)d_in[14];
    const int *writes_dst = (const int *)d_in[15];
    const int *wb_src = (const int *)d_in[16];
    const int *wb_dst = (const int *)d_in[17];

    size_t off = 0;
    unsigned char *q_p   = (unsigned char *)wsoff(d_ws, off, (size_t)NPAPER * CINF);
    unsigned char *q_a   = (unsigned char *)wsoff(d_ws, off, (size_t)NAUTH * CINF);
    unsigned char *q_agg = (unsigned char *)wsoff(d_ws, off, (size_t)NPAPER * CINF);
    float *aggbuf = (float *)wsoff(d_ws, off, (size_t)NPAPER * CINF * 4);
    float *degbuf = (float *)wsoff(d_ws, off, (size_t)NPAPER * 4);
    float *h_p = (float *)wsoff(d_ws, off, (size_t)NPAPER * HDIM * 4);
    float *h_a = (float *)wsoff(d_ws, off, (size_t)NAUTH * HDIM * 4);
    signed char *qwbase = (signed char *)wsoff(d_ws, off, 5 * HDIM * CINF + 5 * COUTD * HDIM);
    float *bscales = (float *)wsoff(d_ws, off, 10 * sizeof(float));
    int *rowsums = (int *)wsoff(d_ws, off, (5 * HDIM + 5 * COUTD) * sizeof(int));
    float *apars = (float *)wsoff(d_ws, off, 10 * 2 * sizeof(float));
    unsigned *mnmx = (unsigned *)wsoff(d_ws, off, 10 * 2 * sizeof(unsigned));

    signed char *qw[10];
    int *rs[10];
    {
        size_t qo = 0, ro = 0;
        for (int i = 0; i < 5; ++i) { qw[i] = qwbase + qo; rs[i] = rowsums + ro; qo += HDIM * CINF; ro += HDIM; }
        for (int i = 5; i < 10; ++i) { qw[i] = qwbase + qo; rs[i] = rowsums + ro; qo += COUTD * HDIM; ro += COUTD; }
    }

    auto run_minmax = [&](const float *x, int n, int slot) {
        minmax_kernel<<<2048, 256, 0, stream>>>(x, n, mnmx + 2 * slot);
    };
    auto run_qact = [&](const float *x, int n, int slot, unsigned char *q) {
        quantize_act_kernel<<<2048, 256, 0, stream>>>(x, n, mnmx + 2 * slot, q, apars + 2 * slot);
    };
    auto run_gemm = [&](const unsigned char *A, int slot, int wi, float *out,
                        int M, int N, int acc, int relu) {
        dim3 grid((M + 63) / 64, N / 64);
        const float *ap = apars + 2 * slot;
        if (!acc)
            gemm_q8_kernel<0, 0><<<grid, 128, 0, stream>>>(A, ap, qw[wi], bscales + wi, rs[wi], out, M, N);
        else if (relu)
            gemm_q8_kernel<1, 1><<<grid, 128, 0, stream>>>(A, ap, qw[wi], bscales + wi, rs[wi], out, M, N);
        else
            gemm_q8_kernel<1, 0><<<grid, 128, 0, stream>>>(A, ap, qw[wi], bscales + wi, rs[wi], out, M, N);
    };
    auto run_agg = [&](const float *feat, const int *src, const int *dst, int ndst,
                       int slot) {
        int n = ndst * CINF;
        zero_kernel<<<(n + 255) / 256, 256, 0, stream>>>(aggbuf, n);
        zero_kernel<<<(ndst + 255) / 256, 256, 0, stream>>>(degbuf, ndst);
        edge_scatter_kernel<<<(NEDGE * 32 + 255) / 256, 256, 0, stream>>>(
            feat, src, dst, aggbuf, degbuf, NEDGE);
        finalize_mean_kernel<<<2048, 256, 0, stream>>>(aggbuf, degbuf, ndst, mnmx + 2 * slot);
        quantize_act_kernel<<<2048, 256, 0, stream>>>(aggbuf, n, mnmx + 2 * slot, q_agg,
                                                      apars + 2 * slot);
    };

    init_slots_kernel<<<1, 32, 0, stream>>>(mnmx, 10);

    for (int i = 0; i < 5; ++i)
        quantize_wt_kernel<<<1, 256, 0, stream>>>(W0[i], HDIM, qw[i], bscales + i, rs[i]);
    for (int i = 0; i < 5; ++i)
        quantize_wt_kernel<<<1, 256, 0, stream>>>(W1[i], COUTD, qw[5 + i], bscales + 5 + i, rs[5 + i]);

    run_minmax(x_p, NPAPER * CINF, 0);
    run_minmax(x_a, NAUTH * CINF, 1);
    run_qact(x_p, NPAPER * CINF, 0, q_p);
    run_qact(x_a, NAUTH * CINF, 1, q_a);

    run_gemm(q_p, 0, 0, h_p, NPAPER, HDIM,  0,  0);
    run_gemm(q_a, 1, 1, h_a, NAUTH, HDIM, 0, 0);

    run_agg(x_p, cites_src, cites_dst, NPAPER, 2);
    run_gemm(q_agg, 2, 2, h_p, NPAPER, HDIM, 1, 0);

    run_agg(x_a, writes_src, writes_dst, NPAPER, 3);
    run_gemm(q_agg, 3, 3, h_p, NPAPER, HDIM, 1,  1);

    run_agg(x_p, wb_src, wb_dst, NAUTH, 4);
    run_gemm(q_agg, 4, 4, h_a, NAUTH, HDIM, 1,  1);

    float *out_p = (float *)d_out;
    float *out_a = out_p + (size_t)NPAPER * COUTD;

    run_minmax(h_p, NPAPER * HDIM, 5);
    run_minmax(h_a, NAUTH * HDIM, 6);
    run_qact(h_p, NPAPER * HDIM, 5, q_p);
    run_qact(h_a, NAUTH * HDIM, 6, q_a);

    run_gemm(q_p, 5, 5, out_p, NPAPER, COUTD, 0, 0);
    run_gemm(q_a, 6, 6, out_a, NAUTH, COUTD, 0, 0);

    run_agg(h_p, cites_src, cites_dst, NPAPER, 7);
    run_gemm(q_agg, 7, 7, out_p, NPAPER, COUTD, 1, 0);

    run_agg(h_a, writes_src, writes_dst, NPAPER, 8);
    run_gemm(q_agg, 8, 8, out_p, NPAPER, COUTD, 1, 0);

    run_agg(h_p, wb_src, wb_dst, NAUTH, 9);
    run_gemm(q_agg, 9, 9, out_a, NAUTH, COUTD, 1, 0);
}
